// Q_ShareScale_VGG16_57964878626994
// MI455X (gfx1250) — 8-bit matrix operand order repaired and checked by reading, compiled, not run on hardware
//
#include <hip/hip_runtime.h>
#include <hip/hip_bf16.h>


typedef __attribute__((ext_vector_type(8))) int v8i;

#define TH_   1.0f
#define LEAK_ 0.5f
#define EPS_  1e-5f
#define QMAX_ 7.0f

__device__ __forceinline__ float slot_scale(const unsigned int* slot) {
    return fmaxf(__uint_as_float(*slot) * (1.0f / QMAX_), 1e-8f);
}


__global__ void zero_f32_k(float* q, size_t n) {
    size_t i = (size_t)blockIdx.x * blockDim.x + threadIdx.x;
    if (i < n) q[i] = 0.0f;
}
__global__ void zero_u32_k(unsigned int* q, int n) {
    int i = blockIdx.x * blockDim.x + threadIdx.x;
    if (i < n) q[i] = 0u;
}


__global__ void wfold_absmax_k(const float* __restrict__ w,
                               const float* __restrict__ g,
                               const float* __restrict__ var,
                               int Ci9, int n, unsigned int* slot) {
    int i = blockIdx.x * blockDim.x + threadIdx.x;
    if (i >= n) return;
    int o = i / Ci9;
    float a = g[o] / sqrtf(var[o] + EPS_);
    atomicMax(slot, __float_as_uint(fabsf(w[i] * a)));
}

__global__ void quantize_w_k(const float* __restrict__ w,
                             const float* __restrict__ g,
                             const float* __restrict__ var,
                             int C, int n, const unsigned int* slot,
                             signed char* __restrict__ wqT) {
    int i = blockIdx.x * blockDim.x + threadIdx.x;
    if (i >= n) return;
    int o   = i / (C * 9);
    int rem = i - o * (C * 9);
    int c   = rem / 9;
    int t   = rem - c * 9;
    float a = g[o] / sqrtf(var[o] + EPS_);
    float s = slot_scale(slot);
    wqT[(size_t)o * (9 * C) + t * C + c] = (signed char)__float2int_rn(w[i] * a / s);
}

__global__ void bfold_absmax_k(const float* g, const float* be, const float* mn,
                               const float* var, int O, unsigned int* slot) {
    int o = blockIdx.x * blockDim.x + threadIdx.x;
    if (o >= O) return;
    float std = sqrtf(var[o] + EPS_);
    float bf  = be[o] - g[o] * mn[o] / std;
    atomicMax(slot, __float_as_uint(fabsf(bf)));
}
__global__ void quantize_b_k(const float* g, const float* be, const float* mn,
                             const float* var, int O, const unsigned int* slot,
                             float* bq) {
    int o = blockIdx.x * blockDim.x + threadIdx.x;
    if (o >= O) return;
    float std = sqrtf(var[o] + EPS_);
    float bf  = be[o] - g[o] * mn[o] / std;
    float s   = slot_scale(slot);
    bq[o] = rintf(bf / s) * s;
}


__global__ void conv0_bn_k(const float* __restrict__ inp,
                           const float* __restrict__ w,
                           const float* g, const float* be,
                           const float* mn, const float* var,
                           float* __restrict__ out, int n) {
    int i = blockIdx.x * blockDim.x + threadIdx.x;
    if (i >= n) return;
    int o = i & 63, x = (i >> 6) & 31, y = (i >> 11) & 31, b = i >> 16;
    float acc = 0.0f;
    for (int c = 0; c < 3; ++c)
        for (int r = 0; r < 3; ++r) {
            int iy = y + r - 1;
            if ((unsigned)iy >= 32u) continue;
            for (int s = 0; s < 3; ++s) {
                int ix = x + s - 1;
                if ((unsigned)ix >= 32u) continue;
                acc += inp[((b * 3 + c) * 32 + iy) * 32 + ix] *
                       w[((o * 3 + c) * 3 + r) * 3 + s];
            }
        }
    float std = sqrtf(var[o] + EPS_);
    out[i] = acc * (g[o] / std) + (be[o] - g[o] * mn[o] / std);
}

__global__ void lif0_k(const float* __restrict__ stat, float* __restrict__ mem,
                       unsigned char* __restrict__ sOut, size_t n) {
    size_t i = (size_t)blockIdx.x * blockDim.x + threadIdx.x;
    if (i >= n) return;
    float u = mem[i] * LEAK_ + stat[i];
    bool sp = (u >= TH_);
    mem[i] = sp ? 0.0f : u;
    int c = (int)(i & 63), x = (int)(i >> 6) & 31, y = (int)(i >> 11) & 31, b = (int)(i >> 16);
    sOut[((size_t)(b * 34 + y + 1) * 34 + x + 1) * 64 + c] = sp ? 1 : 0;
}

__global__ __launch_bounds__(32)
void conv_lif_wmma_k(const unsigned char* __restrict__ sIn,
                     const signed char*  __restrict__ wq,
                     const float*        __restrict__ bq,
                     const unsigned int* __restrict__ wmax_slot,
                     const float*        __restrict__ mem,
                     float*              __restrict__ uTmp,
                     unsigned char*      __restrict__ sOut,
                     unsigned int*       __restrict__ umax_slot,
                     int C, int lgW, int O) {
    const int W  = 1 << lgW;
    const int Wp = W + 2;
    const int lane = threadIdx.x;
    const int half = lane >> 4;
    const int lm   = lane & 15;
    const int K9C  = 9 * C;

    const int p0 = blockIdx.x * 32 + lm;
    const int p1 = p0 + 16;
    const int b0 = p0 >> (2 * lgW), y0 = (p0 >> lgW) & (W - 1), x0 = p0 & (W - 1);
    const int b1 = p1 >> (2 * lgW), y1 = (p1 >> lgW) & (W - 1), x1 = p1 & (W - 1);
    const unsigned char* base0 = sIn + ((size_t)(b0 * Wp + y0) * Wp + x0) * C;
    const unsigned char* base1 = sIn + ((size_t)(b1 * Wp + y1) * Wp + x1) * C;

    const int oB0 = blockIdx.y * 32 + lm;
    const int oB1 = oB0 + 16;
    const signed char* wrow0 = wq + (size_t)oB0 * K9C;
    const signed char* wrow1 = wq + (size_t)oB1 * K9C;

    v8i acc[2][2] = {};

    for (int t = 0; t < 9; ++t) {
        const int dy = (t * 11) >> 5;
        const int dx = t - dy * 3;
        const unsigned char* src0 = base0 + (size_t)(dy * Wp + dx) * C;
        const unsigned char* src1 = base1 + (size_t)(dy * Wp + dx) * C;
        const signed char* wt0 = wrow0 + t * C;
        const signed char* wt1 = wrow1 + t * C;
        if (t < 8) {
            __builtin_prefetch(wt0 + C, 0, 0);
            __builtin_prefetch(wt1 + C, 0, 0);
        }

#pragma unroll 2
        for (int c0 = 0; c0 < C; c0 += 64) {
            v8i a[2], bf[2];
#pragma unroll
            for (int i = 0; i < 8; ++i) {
                const int klA = ((i >> 2) << 5) + ((i & 1) << 2) +
                                (((i >> 1) & 1) << 4) + (half << 3);
                const int klB = klA;
                a[0][i]  = *(const int*)(src0 + c0 + klA);
                a[1][i]  = *(const int*)(src1 + c0 + klA);
                bf[0][i] = *(const int*)(wt0 + c0 + klB);
                bf[1][i] = *(const int*)(wt1 + c0 + klB);
            }
#pragma unroll
            for (int mi = 0; mi < 2; ++mi)
#pragma unroll
                for (int ni = 0; ni < 2; ++ni)
                    acc[mi][ni] = __builtin_amdgcn_wmma_i32_16x16x64_iu8(
                         false, a[mi],  true, bf[ni],
                        acc[mi][ni],  false,  false);
        }
    }

    const float sw = slot_scale(wmax_slot);
    float lmax = 0.0f;
#pragma unroll
    for (int ni = 0; ni < 2; ++ni) {
        const int o = blockIdx.y * 32 + ni * 16 + lm;
        const float bias = bq[o];
#pragma unroll
        for (int mi = 0; mi < 2; ++mi) {
#pragma unroll
            for (int r = 0; r < 8; ++r) {
                const int p = blockIdx.x * 32 + mi * 16 + r + (half << 3);
                const int b = p >> (2 * lgW);
                const int y = (p >> lgW) & (W - 1);
                const int x = p & (W - 1);
                const size_t idx  = (size_t)p * O + o;
                const size_t idxS = ((size_t)(b * Wp + y + 1) * Wp + x + 1) * O + o;
                float xv = (float)acc[mi][ni][r] * sw + bias;
                float u  = mem[idx] * LEAK_ + xv;
                bool sp  = (u >= TH_);
                u = sp ? 0.0f : u;
                uTmp[idx] = u;
                sOut[idxS] = sp ? 1 : 0;
                lmax = fmaxf(lmax, fabsf(u));
            }
        }
    }
    atomicMax(umax_slot, __float_as_uint(lmax));
}

__global__ void quant_mem_k(const float* __restrict__ uTmp,
                            float* __restrict__ mem,
                            const unsigned int* slot, size_t n) {
    size_t i = (size_t)blockIdx.x * blockDim.x + threadIdx.x;
    if (i >= n) return;
    float s = slot_scale(slot);
    mem[i] = rintf(uTmp[i] / s) * s;
}


__global__ void maxpool_u8_k(const unsigned char* __restrict__ in,
                             unsigned char* __restrict__ out,
                             int C, int Hi, int n) {
    int i = blockIdx.x * blockDim.x + threadIdx.x;
    if (i >= n) return;
    int Ho = Hi >> 1;
    int c = i % C;
    int x = (i / C) % Ho;
    int y = (i / (C * Ho)) % Ho;
    int b = i / (C * Ho * Ho);
    int Hpi = Hi + 2, Hpo = Ho + 2;
    const unsigned char* p = in + ((size_t)(b * Hpi + 2 * y + 1) * Hpi + 2 * x + 1) * C + c;
    const int rs = Hpi * C;
    out[((size_t)(b * Hpo + y + 1) * Hpo + x + 1) * C + c] =
        (unsigned char)(p[0] | p[C] | p[rs] | p[rs + C]);
}

__global__ void avgpool_u8_k(const unsigned char* __restrict__ in,
                             float* __restrict__ out, int n) {
    int i = blockIdx.x * blockDim.x + threadIdx.x;
    if (i >= n) return;
    int c = i & 511, b = i >> 9;
    const unsigned char* p = in + (size_t)b * 16 * 512 + c;
    out[i] = 0.25f * (float)(p[5 * 512] + p[6 * 512] + p[9 * 512] + p[10 * 512]);
}

__global__ void fc_k(const float* __restrict__ a,
                     const float* __restrict__ w,
                     const float* __restrict__ b,
                     float* __restrict__ out, int n) {
    int i = blockIdx.x * blockDim.x + threadIdx.x;
    if (i >= n) return;
    int j = i % 10, bb = i / 10;
    float acc = b[j];
    const float* av = a + (size_t)bb * 512;
    const float* wv = w + (size_t)j * 512;
    for (int k = 0; k < 512; ++k) acc += av[k] * wv[k];
    out[i] = acc;
}


extern "C" void kernel_launch(void* const* d_in, const int* in_sizes, int n_in,
                              void* d_out, int out_size, void* d_ws, size_t ws_size,
                              hipStream_t stream) {
    (void)in_sizes; (void)n_in; (void)out_size; (void)ws_size;

    const float* inp = (const float*)d_in[0];
    const float *W_[13], *G_[13], *Be_[13], *Mn_[13], *Vr_[13];
    for (int i = 0; i < 13; ++i) {
        W_[i]  = (const float*)d_in[1 + i];
        G_[i]  = (const float*)d_in[14 + i];
        Be_[i] = (const float*)d_in[27 + i];
        Mn_[i] = (const float*)d_in[40 + i];
        Vr_[i] = (const float*)d_in[53 + i];
    }
    const float* fcw = (const float*)d_in[66];
    const float* fcb = (const float*)d_in[67];
    float* out = (float*)d_out;

    static const int CIN_[13]  = {3, 64, 64, 128, 128, 256, 256, 256, 512, 512, 512, 512, 512};
    static const int COUT_[13] = {64, 64, 128, 128, 256, 256, 256, 512, 512, 512, 512, 512, 512};
    static const int HH_[13]   = {32, 32, 16, 16, 8, 8, 8, 4, 4, 4, 2, 2, 2};
    const int B = 128;

    char* p = (char*)d_ws;
    auto carve = [&](size_t bytes) -> void* {
        void* r = (void*)p;
        p += (bytes + 255) & ~(size_t)255;
        return r;
    };
    const size_t n0 = (size_t)B * 64 * 32 * 32;
    float* stat = (float*)carve(n0 * 4);

    size_t memElems[13], memOff[13], totMem = 0;
    for (int j = 0; j < 13; ++j) {
        memElems[j] = (size_t)B * COUT_[j] * HH_[j] * HH_[j];
        memOff[j] = totMem;
        totMem += memElems[j];
    }
    float* memBase = (float*)carve(totMem * 4);
    float* mem[13];
    for (int j = 0; j < 13; ++j) mem[j] = memBase + memOff[j];

    float* uTmp = (float*)carve(n0 * 4);

    char* spkStart = p;
    auto spk = [&](int H, int C) -> unsigned char* {
        return (unsigned char*)carve((size_t)B * (H + 2) * (H + 2) * C);
    };
    unsigned char* spL0 = spk(32, 64);
    unsigned char* spCv[13];
    for (int j = 1; j < 13; ++j) spCv[j] = spk(HH_[j], COUT_[j]);
    unsigned char* spP1 = spk(16, 64);
    unsigned char* spP2 = spk(8, 128);
    unsigned char* spP3 = spk(4, 256);
    unsigned char* spP4 = spk(2, 512);
    size_t spkBytes = (size_t)(p - spkStart);

    signed char* wq[13];
    float* bq[13];
    for (int j = 1; j < 13; ++j) {
        wq[j] = (signed char*)carve((size_t)COUT_[j] * CIN_[j] * 9);
        bq[j] = (float*)carve((size_t)COUT_[j] * 4);
    }
    unsigned int* wmax = (unsigned int*)carve(13 * 4);
    unsigned int* bmax = (unsigned int*)carve(13 * 4);
    unsigned int* umax = (unsigned int*)carve(13 * 4);
    float* avg = (float*)carve((size_t)B * 512 * 4);

    zero_f32_k<<<(unsigned)((totMem + 255) / 256), 256, 0, stream>>>(memBase, totMem);
    zero_f32_k<<<(unsigned)((spkBytes / 4 + 255) / 256), 256, 0, stream>>>((float*)spkStart, spkBytes / 4);
    zero_u32_k<<<1, 32, 0, stream>>>(wmax, 13);
    zero_u32_k<<<1, 32, 0, stream>>>(bmax, 13);

    conv0_bn_k<<<(unsigned)((n0 + 255) / 256), 256, 0, stream>>>(
        inp, W_[0], G_[0], Be_[0], Mn_[0], Vr_[0], stat, (int)n0);

    for (int j = 1; j < 13; ++j) {
        int C   = CIN_[j];
        int nw  = COUT_[j] * C * 9;
        wfold_absmax_k<<<(nw + 255) / 256, 256, 0, stream>>>(W_[j], G_[j], Vr_[j], C * 9, nw, wmax + j);
        quantize_w_k<<<(nw + 255) / 256, 256, 0, stream>>>(W_[j], G_[j], Vr_[j], C, nw, wmax + j, wq[j]);
        bfold_absmax_k<<<(COUT_[j] + 63) / 64, 64, 0, stream>>>(G_[j], Be_[j], Mn_[j], Vr_[j], COUT_[j], bmax + j);
        quantize_b_k<<<(COUT_[j] + 63) / 64, 64, 0, stream>>>(G_[j], Be_[j], Mn_[j], Vr_[j], COUT_[j], bmax + j, bq[j]);
    }

    auto CONV = [&](int j, const unsigned char* in, unsigned char* sOut) {
        int H = HH_[j], C = CIN_[j], O = COUT_[j];
        int lgW = 31 - __builtin_clz((unsigned)H);
        dim3 grid((unsigned)((B * H * H) / 32), (unsigned)(O / 32));
        conv_lif_wmma_k<<<grid, 32, 0, stream>>>(in, wq[j], bq[j], wmax + j,
                                                 mem[j], uTmp, sOut, umax + j,
                                                 C, lgW, O);
        size_t n = memElems[j];
        quant_mem_k<<<(unsigned)((n + 255) / 256), 256, 0, stream>>>(uTmp, mem[j], umax + j, n);
    };
    auto MAXP = [&](const unsigned char* in, unsigned char* o, int C, int Hi) {
        int n = B * C * (Hi / 2) * (Hi / 2);
        maxpool_u8_k<<<(n + 255) / 256, 256, 0, stream>>>(in, o, C, Hi, n);
    };

    for (int t = 0; t < 4; ++t) {
        zero_u32_k<<<1, 32, 0, stream>>>(umax, 13);
        lif0_k<<<(unsigned)((n0 + 255) / 256), 256, 0, stream>>>(stat, mem[0], spL0, n0);

        CONV(1, spL0, spCv[1]);       MAXP(spCv[1], spP1, 64, 32);
        CONV(2, spP1, spCv[2]);
        CONV(3, spCv[2], spCv[3]);    MAXP(spCv[3], spP2, 128, 16);
        CONV(4, spP2, spCv[4]);
        CONV(5, spCv[4], spCv[5]);
        CONV(6, spCv[5], spCv[6]);    MAXP(spCv[6], spP3, 256, 8);
        CONV(7, spP3, spCv[7]);
        CONV(8, spCv[7], spCv[8]);
        CONV(9, spCv[8], spCv[9]);    MAXP(spCv[9], spP4, 512, 4);
        CONV(10, spP4, spCv[10]);
        CONV(11, spCv[10], spCv[11]);
        CONV(12, spCv[11], spCv[12]);

        int na = B * 512;
        avgpool_u8_k<<<(na + 255) / 256, 256, 0, stream>>>(spCv[12], avg, na);
        int nf = B * 10;
        fc_k<<<(nf + 255) / 256, 256, 0, stream>>>(avg, fcw, fcb, out + (size_t)t * nf, nf);
    }
}
